// MSDeformAttn_Fusion_45973329937036
// MI455X (gfx1250) — matrix operand order repaired and checked by reading, not compiled, not run on hardware
//
#include <hip/hip_runtime.h>
#include <hip/hip_bf16.h>

// ---------------------------------------------------------------------------
// Flow-guided multi-scale deformable attention, CDNA5 (gfx1250) WMMA version.
// B=2, T=3, C=256, H=W=64, M=8 heads, P=4 points, DH=32.
// Convs run as implicit GEMM on v_wmma_f32_16x16x32_bf16.
// ---------------------------------------------------------------------------

typedef __attribute__((ext_vector_type(16))) __bf16 v16bf;
typedef __attribute__((ext_vector_type(8)))  float  v8f;

__device__ __forceinline__ unsigned short f2bf(float f) {
  unsigned int u = __float_as_uint(f);
  u += 0x7FFFu + ((u >> 16) & 1u);           // round-to-nearest-even
  return (unsigned short)(u >> 16);
}

// ---------------------- elementwise f32 -> bf16 cast -----------------------
__global__ void cast_bf16_kernel(const float* __restrict__ in,
                                 unsigned short* __restrict__ out, int n) {
  int i = blockIdx.x * blockDim.x + threadIdx.x;
  if (i < n) out[i] = f2bf(in[i]);
}

// --------- weight transform: OIHW f32 -> [ktap][ci][co] bf16 ----------------
__global__ void weight_prep_kernel(const float* __restrict__ w,
                                   unsigned short* __restrict__ w2,
                                   int Cin, int Cout) {
  int n = 9 * Cin * Cout;
  int idx = blockIdx.x * blockDim.x + threadIdx.x;
  if (idx >= n) return;
  int co   = idx % Cout;
  int r    = idx / Cout;
  int ci   = r % Cin;
  int kpos = r / Cin;                         // ky*3+kx
  w2[idx] = f2bf(w[(co * Cin + ci) * 9 + kpos]);
}

// ---------------------------------------------------------------------------
// 3x3 SAME conv, implicit GEMM on WMMA bf16.
// Block = 256 threads (8 wave32s). Block tile: 8x8 pixels x 64 cout.
// Wave (pg,cg): 16 pixels (rows 2pg,2pg+1 of tile) x 32 cout -> 2 accumulators.
// Inner loop: ci chunks of 32; per chunk stage 10x10x32 input halo + 9x32x64
// weights in LDS, then 9 kernel-tap WMMAs per accumulator.
// ---------------------------------------------------------------------------
__global__ __launch_bounds__(256) void conv3x3_wmma_kernel(
    const unsigned short* __restrict__ in,   // (Nimg, Cin, 64, 64) bf16
    const unsigned short* __restrict__ w2,   // (9, Cin, Cout)      bf16
    const float* __restrict__ bias,          // (Cout)
    float* __restrict__ out,                 // (Nimg, Cout, 64, 64)
    int Cin, int Cout) {
  __shared__ __align__(16) unsigned short ilds[100 * 32];     // [pix10x10][ci32]
  __shared__ __align__(16) unsigned short wlds[9 * 64 * 32];  // [ktap][co64][ci32]

  const int tid  = threadIdx.x;
  const int lane = tid & 31;
  const int wave = tid >> 5;
  const int pg   = wave >> 1;                // 0..3 pixel group
  const int cg   = wave & 1;                 // 0..1 channel group
  const int half = lane >> 4;
  const int n16  = lane & 15;

  const int img   = blockIdx.z;
  const int co0   = blockIdx.y * 64;
  const int tileY = (blockIdx.x >> 3) * 8;
  const int tileX = (blockIdx.x & 7) * 8;

  // A-matrix row (= pixel) for this lane (same for both lane halves)
  const int pl = pg * 16 + n16;
  const int yl = pl >> 3, xl = pl & 7;

  const int coA = co0 + cg * 32 + n16;
  const int coB = coA + 16;
  const float bA = (coA < Cout) ? bias[coA] : 0.f;
  const float bB = (coB < Cout) ? bias[coB] : 0.f;
  v8f acc0, acc1;
  for (int r = 0; r < 8; ++r) { acc0[r] = bA; acc1[r] = bB; }

  for (int ci0 = 0; ci0 < Cin; ci0 += 32) {
    // ---- stage input halo tile (zero-padded) ----
    for (int idx = tid; idx < 100 * 32; idx += 256) {
      int i   = idx / 100;                   // ci local
      int pix = idx % 100;
      int yy = pix / 10, xx = pix % 10;
      int gy = tileY - 1 + yy, gx = tileX - 1 + xx;
      unsigned short v = 0;
      if (gy >= 0 && gy < 64 && gx >= 0 && gx < 64)
        v = in[(((size_t)img * Cin + ci0 + i) << 12) + (gy << 6) + gx];
      ilds[pix * 32 + i] = v;
    }
    // ---- stage weights, transposed to [co][ci] for contiguous B K-reads ----
    for (int idx = tid; idx < 9 * 32 * 64; idx += 256) {
      int j    = idx & 63;                   // co local
      int r    = idx >> 6;
      int i    = r & 31;                     // ci local
      int kpos = r >> 5;
      unsigned short v = 0;
      if (co0 + j < Cout)
        v = w2[((size_t)kpos * Cin + ci0 + i) * Cout + co0 + j];
      wlds[(kpos * 64 + j) * 32 + i] = v;
    }
    __syncthreads();

    #pragma unroll
    for (int kpos = 0; kpos < 9; ++kpos) {
      const int ky = kpos / 3, kx = kpos % 3;
      // A fragment: 16 pixels x 32 ci (K = half*16 + j, contiguous in LDS)
      const unsigned short* abase = &ilds[((yl + ky) * 10 + xl + kx) * 32];
      v16bf aF;
      {
        uint4* ap = reinterpret_cast<uint4*>(&aF);
        ap[0] = *reinterpret_cast<const uint4*>(abase + half * 16);
        ap[1] = *reinterpret_cast<const uint4*>(abase + half * 16 + 8);
      }
      // B fragments: 32 ci x 16 co (K = half*16 + j, contiguous in LDS)
      const unsigned short* b0 = &wlds[(kpos * 64 + cg * 32 + n16) * 32 + half * 16];
      const unsigned short* b1 = &wlds[(kpos * 64 + cg * 32 + 16 + n16) * 32 + half * 16];
      v16bf bF0, bF1;
      {
        uint4* bp = reinterpret_cast<uint4*>(&bF0);
        bp[0] = *reinterpret_cast<const uint4*>(b0);
        bp[1] = *reinterpret_cast<const uint4*>(b0 + 8);
        uint4* bq = reinterpret_cast<uint4*>(&bF1);
        bq[0] = *reinterpret_cast<const uint4*>(b1);
        bq[1] = *reinterpret_cast<const uint4*>(b1 + 8);
      }
      acc0 = __builtin_amdgcn_wmma_f32_16x16x32_bf16(false, aF, false, bF0,
                                                     (short)0, acc0, false, false);
      acc1 = __builtin_amdgcn_wmma_f32_16x16x32_bf16(false, aF, false, bF1,
                                                     (short)0, acc1, false, false);
    }
    __syncthreads();
  }

  // D layout: element r -> pixel row m = r + 8*half; lane writes one row of 8 x
  const int y  = tileY + pg * 2 + half;
  const int x0 = tileX;
  if (coA < Cout) {
    float* p = &out[(((size_t)img * Cout + coA) << 12) + (y << 6) + x0];
    *reinterpret_cast<float4*>(p)     = make_float4(acc0[0], acc0[1], acc0[2], acc0[3]);
    *reinterpret_cast<float4*>(p + 4) = make_float4(acc0[4], acc0[5], acc0[6], acc0[7]);
  }
  if (coB < Cout) {
    float* p = &out[(((size_t)img * Cout + coB) << 12) + (y << 6) + x0];
    *reinterpret_cast<float4*>(p)     = make_float4(acc1[0], acc1[1], acc1[2], acc1[3]);
    *reinterpret_cast<float4*>(p + 4) = make_float4(acc1[4], acc1[5], acc1[6], acc1[7]);
  }
}

// ---------------------------------------------------------------------------
// Flow-guided deformable sampling. One wave per (b, q, head); lane = head dim.
// Softmax over T*P=12 logits, flow-added offsets, 4-tap bilinear, bf16 NCHW out.
// ---------------------------------------------------------------------------
__global__ __launch_bounds__(256) void deform_sample_kernel(
    const float* __restrict__ valueF,        // (B,T,256,64,64)
    const float* __restrict__ offF,          // (B,192,64,64)
    const float* __restrict__ attF,          // (B,96,64,64)
    const float* __restrict__ refp,          // (B,4096,3,2)
    const float* __restrict__ flowF,         // (B,2,2,64,64)
    const float* __restrict__ flowB,         // (B,2,2,64,64)
    unsigned short* __restrict__ outBf) {    // (B,256,64,64) bf16
  const int wid  = blockIdx.x * 8 + (threadIdx.x >> 5);
  const int lane = threadIdx.x & 31;         // head-dim channel d
  const int m = wid & 7;
  const int q = (wid >> 3) & 4095;
  const int b = wid >> 15;

  // softmax over the 12 (t,p) attention logits for this (b,q,m)
  float a[12];
  float mx = -1e30f;
  #pragma unroll
  for (int i = 0; i < 12; ++i) {
    a[i] = attF[(((size_t)b * 96 + m * 12 + i) << 12) + q];
    mx = fmaxf(mx, a[i]);
  }
  float s = 0.f;
  #pragma unroll
  for (int i = 0; i < 12; ++i) { a[i] = __expf(a[i] - mx); s += a[i]; }
  const float inv = 1.f / s;

  float acc = 0.f;
  #pragma unroll
  for (int t = 0; t < 3; ++t) {
    const float rx = refp[(((size_t)b * 4096 + q) * 3 + t) * 2 + 0];
    const float ry = refp[(((size_t)b * 4096 + q) * 3 + t) * 2 + 1];
    // flow stack: t=0 <- backward[:,0], t=1 <- 0, t=2 <- forward[:,1]
    float fx = 0.f, fy = 0.f;
    if (t == 0) {
      fx = flowB[(((size_t)b * 4 + 0) << 12) + q];
      fy = flowB[(((size_t)b * 4 + 1) << 12) + q];
    } else if (t == 2) {
      fx = flowF[(((size_t)b * 4 + 2) << 12) + q];
      fy = flowF[(((size_t)b * 4 + 3) << 12) + q];
    }
    const float* vbase = valueF + ((((size_t)b * 3 + t) * 256 + m * 32 + lane) << 12);
    #pragma unroll
    for (int p = 0; p < 4; ++p) {
      const int ch = ((m * 3 + t) * 4 + p) * 2;
      const float ox = offF[(((size_t)b * 192 + ch) << 12) + q] + fx;
      const float oy = offF[(((size_t)b * 192 + ch + 1) << 12) + q] + fy;
      const float lx = rx + ox * (1.f / 64.f);
      const float ly = ry + oy * (1.f / 64.f);
      const float x = lx * 64.f - 0.5f;
      const float y = ly * 64.f - 0.5f;
      const float x0f = floorf(x), y0f = floorf(y);
      const float wx = x - x0f, wy = y - y0f;
      const int x0 = (int)x0f, y0 = (int)y0f;
      float smp = 0.f;
      if (x0 >= 0 && x0 < 64 && y0 >= 0 && y0 < 64)
        smp += (1.f - wx) * (1.f - wy) * vbase[(y0 << 6) + x0];
      if (x0 + 1 >= 0 && x0 + 1 < 64 && y0 >= 0 && y0 < 64)
        smp += wx * (1.f - wy) * vbase[(y0 << 6) + x0 + 1];
      if (x0 >= 0 && x0 < 64 && y0 + 1 >= 0 && y0 + 1 < 64)
        smp += (1.f - wx) * wy * vbase[((y0 + 1) << 6) + x0];
      if (x0 + 1 >= 0 && x0 + 1 < 64 && y0 + 1 >= 0 && y0 + 1 < 64)
        smp += wx * wy * vbase[((y0 + 1) << 6) + x0 + 1];
      acc += (a[t * 4 + p] * inv) * smp;
    }
  }
  outBf[(((size_t)b * 256 + m * 32 + lane) << 12) + q] = f2bf(acc);
}

// ---------------------------------------------------------------------------
extern "C" void kernel_launch(void* const* d_in, const int* in_sizes, int n_in,
                              void* d_out, int out_size, void* d_ws, size_t ws_size,
                              hipStream_t stream) {
  (void)in_sizes; (void)n_in; (void)out_size; (void)ws_size;

  const float* query   = (const float*)d_in[0];   // (2,3,256,64,64)
  const float* refp    = (const float*)d_in[1];   // (2,4096,3,2)
  const float* inflat  = (const float*)d_in[2];   // (2,3,256,64,64)
  const float* flowF   = (const float*)d_in[6];   // (2,2,2,64,64)
  const float* flowB   = (const float*)d_in[7];   // (2,2,2,64,64)
  const float* w_value = (const float*)d_in[8];
  const float* b_value = (const float*)d_in[9];
  const float* w_off   = (const float*)d_in[10];
  const float* b_off   = (const float*)d_in[11];
  const float* w_attn  = (const float*)d_in[12];
  const float* b_attn  = (const float*)d_in[13];
  const float* w_out   = (const float*)d_in[14];
  const float* b_out   = (const float*)d_in[15];
  float* out = (float*)d_out;                     // (2,256,64,64)

  // workspace carve-up (256B aligned), ~70 MB total
  size_t pos = 0;
  char* base = (char*)d_ws;
  auto carve = [&](size_t bytes) -> void* {
    void* p = base + pos;
    pos += (bytes + 255) & ~(size_t)255;
    return p;
  };
  unsigned short* qbf  = (unsigned short*)carve(2ull * 768 * 4096 * 2);  // query bf16
  unsigned short* fbf  = (unsigned short*)carve(6ull * 256 * 4096 * 2);  // input_flatten bf16
  unsigned short* wv2  = (unsigned short*)carve(9ull * 256 * 256 * 2);
  unsigned short* wo2  = (unsigned short*)carve(9ull * 768 * 192 * 2);
  unsigned short* wa2  = (unsigned short*)carve(9ull * 768 * 96 * 2);
  unsigned short* wt2  = (unsigned short*)carve(9ull * 256 * 256 * 2);
  float*          valF = (float*)carve(6ull * 256 * 4096 * 4);
  float*          offF = (float*)carve(2ull * 192 * 4096 * 4);
  float*          attF = (float*)carve(2ull * 96 * 4096 * 4);
  unsigned short* aout = (unsigned short*)carve(2ull * 256 * 4096 * 2);

  const int nTensor = 2 * 768 * 4096;  // == 6*256*4096
  cast_bf16_kernel<<<(nTensor + 255) / 256, 256, 0, stream>>>(query, qbf, nTensor);
  cast_bf16_kernel<<<(nTensor + 255) / 256, 256, 0, stream>>>(inflat, fbf, nTensor);

  weight_prep_kernel<<<(9 * 256 * 256 + 255) / 256, 256, 0, stream>>>(w_value, wv2, 256, 256);
  weight_prep_kernel<<<(9 * 768 * 192 + 255) / 256, 256, 0, stream>>>(w_off,   wo2, 768, 192);
  weight_prep_kernel<<<(9 * 768 * 96  + 255) / 256, 256, 0, stream>>>(w_attn,  wa2, 768, 96);
  weight_prep_kernel<<<(9 * 256 * 256 + 255) / 256, 256, 0, stream>>>(w_out,   wt2, 256, 256);

  // value conv: 6 images, 256->256
  conv3x3_wmma_kernel<<<dim3(64, 4, 6), 256, 0, stream>>>(fbf, wv2, b_value, valF, 256, 256);
  // offset conv: 2 images, 768->192
  conv3x3_wmma_kernel<<<dim3(64, 3, 2), 256, 0, stream>>>(qbf, wo2, b_off, offF, 768, 192);
  // attention conv: 2 images, 768->96
  conv3x3_wmma_kernel<<<dim3(64, 2, 2), 256, 0, stream>>>(qbf, wa2, b_attn, attF, 768, 96);

  // deformable sampling: 2*4096*8 waves / 8 waves per block
  deform_sample_kernel<<<8192, 256, 0, stream>>>(valF, offF, attF, refp, flowF, flowB, aout);

  // output conv: 2 images, 256->256, fp32 result straight to d_out
  conv3x3_wmma_kernel<<<dim3(64, 4, 2), 256, 0, stream>>>(aout, wt2, b_out, out, 256, 256);
}
